// AssembleModel_68169720922507
// MI455X (gfx1250) — matrix operand order repaired and checked by reading, not compiled, not run on hardware
//
#include <hip/hip_runtime.h>
#include <hip/hip_bf16.h>
#include <cstddef>

// ---------------------------------------------------------------------------
// Problem constants (from reference)
// ---------------------------------------------------------------------------
#define BS_   1024
#define NN_   10
#define NNODE (BS_ * NN_)      // 10240
#define CL_   3072             // C*L
#define D_    384
#define K2_   768              // 2*D
#define HID_  512
#define E_TOT 92160            // BS * NN * (NN-1)
#define R_    1000.0f

typedef __attribute__((ext_vector_type(16))) __bf16        v16bf;
typedef __attribute__((ext_vector_type(8)))  float         v8f;
typedef __attribute__((ext_vector_type(4)))  float         f32x4;
typedef __attribute__((ext_vector_type(4)))  unsigned int  v4u;
typedef __attribute__((ext_vector_type(4)))  unsigned int  u32x4;
typedef __attribute__((ext_vector_type(8)))  int           i32x8;
typedef __attribute__((ext_vector_type(4)))  int           i32x4;

#define AS1 __attribute__((address_space(1)))
#define AS3 __attribute__((address_space(3)))
typedef unsigned long long u64_;

union FragBF { v16bf v; v4u q[2]; unsigned int u[8]; };
union BF2    { __bf16 b[2]; unsigned int u; };

// ---------------------------------------------------------------------------
// CDNA5 data-movement feature detection (device pass only)
// ---------------------------------------------------------------------------
#if defined(__gfx1250__) && __has_builtin(__builtin_amdgcn_global_load_async_to_lds_b128)
#define HAVE_ASYNC_LDS 1
#else
#define HAVE_ASYNC_LDS 0
#endif

#if defined(__gfx1250__) && __has_builtin(__builtin_amdgcn_tensor_load_to_lds)
#define HAVE_TDM 1
#else
#define HAVE_TDM 0
#endif

// 16-byte global -> LDS copy; async DMA when available, sync fallback otherwise.
__device__ __forceinline__ void async_copy16(const void* g, void* l) {
#if HAVE_ASYNC_LDS
  __builtin_amdgcn_global_load_async_to_lds_b128(
      (AS1 i32x4*)(u64_)g, (AS3 i32x4*)(unsigned)(u64_)l, 0, 0);
#else
  *(v4u*)l = *(const v4u*)g;
#endif
}

__device__ __forceinline__ void wait_async() {
#if HAVE_ASYNC_LDS
#if __has_builtin(__builtin_amdgcn_s_wait_asynccnt)
  __builtin_amdgcn_s_wait_asynccnt(0);
#else
  asm volatile("s_wait_asynccnt 0" ::: "memory");
#endif
#endif
}

__device__ __forceinline__ void wait_tensor() {
#if __has_builtin(__builtin_amdgcn_s_wait_tensorcnt)
  __builtin_amdgcn_s_wait_tensorcnt(0);
#else
  asm volatile("s_wait_tensorcnt 0" ::: "memory");
#endif
}

#if HAVE_TDM
// 1-D contiguous TDM transfer: n_u32 dwords from global -> LDS.
// D# per CDNA5 ISA ch.8: group0 = {count|flags, lds_addr, global_addr lo/hi|type},
// group1 carries data_size / tensor_dim0 / tile_dim0 / stride packing.
__device__ __forceinline__ void tdm_load_1d(unsigned lds_off, u64_ gaddr, unsigned n_u32) {
  u32x4 g0;
  g0[0] = 1u;                                              // count=1, user mode
  g0[1] = lds_off;                                         // lds_addr (bytes)
  g0[2] = (unsigned)(gaddr & 0xFFFFFFFFull);               // global_addr[31:0]
  g0[3] = (unsigned)((gaddr >> 32) & 0x01FFFFFFull)        // global_addr[56:32]
        | 0x80000000u;                                     // type=2 (bits 127:126)
  i32x8 g1;
  g1[0] = (int)(2u << 16);                                 // data_size = 4 bytes
  g1[1] = (int)((n_u32 & 0xFFFFu) << 16);                  // tensor_dim0[15:0]
  g1[2] = (int)(((n_u32 >> 16) & 0xFFFFu) | (1u << 16));   // dim0 hi | tensor_dim1=1
  g1[3] = (int)((n_u32 & 0xFFFFu) << 16);                  // tile_dim0 (<=65535)
  g1[4] = 0;                                               // tile_dim1/2 unused
  g1[5] = (int)n_u32;                                      // tensor_dim0_stride lo
  g1[6] = 0;
  g1[7] = 0;
  i32x4 z4 = {0, 0, 0, 0};
  i32x8 z8 = {0, 0, 0, 0, 0, 0, 0, 0};
  __builtin_amdgcn_tensor_load_to_lds(g0, g1, z4, z4, z8, 0);
}
#endif

// ---------------------------------------------------------------------------
// WMMA fragment loaders (bf16, 16x16x32).
// A frag: lane holds row; K elements [16g,16g+16) -> 2x b128.
// B frag: K-pair packed LDS (u32 idx = kpair*bn + n) -> 8x ds_load_b32.
// ---------------------------------------------------------------------------
__device__ __forceinline__ v16bf load_a_frag(const __bf16* rowk, int g) {
  const v4u* p = (const v4u*)rowk;
  FragBF f;
  f.q[0] = p[2 * g];
  f.q[1] = p[2 * g + 1];
  return f.v;
}

__device__ __forceinline__ v16bf load_b_frag(const __bf16* slab, int bn, int col, int g) {
  const unsigned int* p = (const unsigned int*)slab;
  FragBF f;
#pragma unroll
  for (int i = 0; i < 8; ++i) f.u[i] = p[(i + 8 * g) * bn + col];
  return f.v;
}

// ---------------------------------------------------------------------------
// Kernel 0: one-shot weight conversion f32 -> K-pair-packed bf16.
//   packed u32 index = (k>>1)*N + n ; low half = even k, high half = odd k.
//   W2 is zero-padded from 17 to 32 columns for the WMMA GEMM2.
// ---------------------------------------------------------------------------
__global__ __launch_bounds__(256) void pack_weights(
    const float* __restrict__ encW, const float* __restrict__ W1,
    const float* __restrict__ W2,
    unsigned* __restrict__ encWp, unsigned* __restrict__ W1p,
    unsigned* __restrict__ W2p)
{
  const int NT = gridDim.x * 256;
  const int tid = blockIdx.x * 256 + threadIdx.x;

  for (int m = tid; m < (CL_ / 2) * D_; m += NT) {
    int kp = m / D_, n = m % D_;
    BF2 u;
    u.b[0] = (__bf16)encW[(size_t)(2 * kp) * D_ + n];
    u.b[1] = (__bf16)encW[(size_t)(2 * kp + 1) * D_ + n];
    encWp[m] = u.u;
  }
  for (int m = tid; m < (K2_ / 2) * HID_; m += NT) {
    int kp = m / HID_, n = m % HID_;
    BF2 u;
    u.b[0] = (__bf16)W1[(size_t)(2 * kp) * HID_ + n];
    u.b[1] = (__bf16)W1[(size_t)(2 * kp + 1) * HID_ + n];
    W1p[m] = u.u;
  }
  for (int m = tid; m < (HID_ / 2) * 32; m += NT) {
    int kp = m / 32, n = m % 32;
    float v0 = (n < 17) ? W2[(size_t)(2 * kp) * 17 + n] : 0.0f;
    float v1 = (n < 17) ? W2[(size_t)(2 * kp + 1) * 17 + n] : 0.0f;
    BF2 u;
    u.b[0] = (__bf16)v0;
    u.b[1] = (__bf16)v1;
    W2p[m] = u.u;
  }
}

// ---------------------------------------------------------------------------
// Kernel 1: x = relu(img @ enc_W + enc_b) -> bf16.
// Block tile 128x64, 8 waves, wave = 32x32 (2x2 WMMA), BK=32.
// A: vectorized f32 b128 loads + fused bf16 cvt (img is the 126MB stream).
// B: async b128 DMA from pre-packed bf16 weights (16B per lane per step).
// ---------------------------------------------------------------------------
__global__ __launch_bounds__(256) void enc_gemm_relu(
    const float* __restrict__ img, const unsigned* __restrict__ encWp,
    const float* __restrict__ bias, __bf16* __restrict__ xout)
{
  __shared__ alignas(16) __bf16 As[128 * 32];
  __shared__ alignas(16) __bf16 Bs[32 * 64];   // K-pair packed [16][64] u32

  const int t    = threadIdx.x;
  const int lane = t & 31;
  const int wave = t >> 5;
  const int row0 = blockIdx.y * 128;
  const int col0 = blockIdx.x * 64;
  const int wm = wave >> 1, wn = wave & 1;
  const int g = lane >> 4, ln = lane & 15;

  v8f acc[2][2] = {};

  for (int kk = 0; kk < CL_; kk += 32) {
    // B: one 16-byte async transfer per thread (4 KB slab)
    {
      int kp = t >> 4, ns = (t & 15) * 4;
      const unsigned* gsrc = encWp + (size_t)(kk / 2 + kp) * D_ + col0 + ns;
      async_copy16(gsrc, (char*)Bs + (kp * 64 + ns) * 4);
    }
    // A: 128x32 f32 -> bf16, b128 global loads
#pragma unroll
    for (int i = 0; i < 4; ++i) {
      int idx = (t + i * 256) * 4;          // element index 0..4095
      int r = idx >> 5, k = idx & 31;
      f32x4 v = *(const f32x4*)(img + (size_t)(row0 + r) * CL_ + kk + k);
      BF2 lo, hi;
      lo.b[0] = (__bf16)v.x; lo.b[1] = (__bf16)v.y;
      hi.b[0] = (__bf16)v.z; hi.b[1] = (__bf16)v.w;
      unsigned* d = (unsigned*)(As + idx);
      d[0] = lo.u; d[1] = hi.u;
    }
    wait_async();
    __syncthreads();

    v16bf a0 = load_a_frag(As + (wm * 32 + ln) * 32,      g);
    v16bf a1 = load_a_frag(As + (wm * 32 + 16 + ln) * 32, g);
    v16bf b0 = load_b_frag(Bs, 64, wn * 32 + ln,      g);
    v16bf b1 = load_b_frag(Bs, 64, wn * 32 + 16 + ln, g);

    acc[0][0] = __builtin_amdgcn_wmma_f32_16x16x32_bf16(false, a0, false, b0, (short)0, acc[0][0], false, false);
    acc[0][1] = __builtin_amdgcn_wmma_f32_16x16x32_bf16(false, a0, false, b1, (short)0, acc[0][1], false, false);
    acc[1][0] = __builtin_amdgcn_wmma_f32_16x16x32_bf16(false, a1, false, b0, (short)0, acc[1][0], false, false);
    acc[1][1] = __builtin_amdgcn_wmma_f32_16x16x32_bf16(false, a1, false, b1, (short)0, acc[1][1], false, false);
    __syncthreads();
  }

#pragma unroll
  for (int ti = 0; ti < 2; ++ti)
#pragma unroll
    for (int tj = 0; tj < 2; ++tj) {
      int c = col0 + wn * 32 + tj * 16 + ln;
      float bv = bias[c];
#pragma unroll
      for (int i = 0; i < 8; ++i) {
        int r = row0 + wm * 32 + ti * 16 + i + 8 * g;
        float v = acc[ti][tj][i] + bv;
        v = v > 0.0f ? v : 0.0f;
        xout[(size_t)r * D_ + c] = (__bf16)v;
      }
    }
}

// ---------------------------------------------------------------------------
// Kernel 2: per 64-edge block, fully fused:
//   h     = relu(gather([x[dst],x[src]]) @ W1 + b1)   (WMMA; A gathered by
//           per-lane async b128 DMA, W1 slab by TDM 1-D descriptor)
//   preds = h @ W2pad + b2                            (WMMA from LDS h tile)
//   eigh(sym4(preds[6:16])) -> quat; mask; exp(); all outputs
// ---------------------------------------------------------------------------
__global__ __launch_bounds__(256) void msg_edge_kernel(
    const __bf16* __restrict__ xbf, const float* __restrict__ pos,
    const unsigned* __restrict__ W1p, const float* __restrict__ b1,
    const unsigned* __restrict__ W2p, const float* __restrict__ b2,
    float* __restrict__ out)
{
  __shared__ alignas(16) __bf16 As[64 * 32];     //  4 KB
  __shared__ alignas(16) __bf16 Bs[32 * 512];    // 32 KB (W1 slab, later W2)
  __shared__ alignas(16) __bf16 Hs[64 * 512];    // 64 KB
  __shared__ float  Ps[64 * 20];
  __shared__ int    sSrc[64], sDst[64];

  const int t    = threadIdx.x;
  const int lane = t & 31;
  const int wave = t >> 5;
  const int e0   = blockIdx.x * 64;
  const int wm = wave & 3;        // m-tile (16 edges)
  const int nh = wave >> 2;       // GEMM1 column half
  const int g = lane >> 4, ln = lane & 15;

  if (t < 64) {
    int e = e0 + t;
    int b = e / 90, p = e % 90;
    int i = p / 9, q = p % 9;
    int j = q + (q >= i ? 1 : 0);
    sDst[t] = b * NN_ + i;
    sSrc[t] = b * NN_ + j;
  }
  __syncthreads();

  v8f acc[16] = {};

  for (int kk = 0; kk < K2_; kk += 32) {
    // A: gathered 64x32 bf16 -- exactly one async b128 per lane
    {
      int r = t >> 2, seg = (t & 3) * 8;
      int kg = kk + seg;
      int node = (kg < D_) ? sDst[r] : sSrc[r];
      int col  = (kg < D_) ? kg : kg - D_;
      async_copy16(xbf + (size_t)node * D_ + col, As + r * 32 + seg);
    }
    // B: contiguous 32 KB packed-W1 slab
#if HAVE_TDM
    if (wave == 0) {
      tdm_load_1d((unsigned)(u64_)(void*)Bs,
                  (u64_)(const void*)(W1p + (size_t)(kk / 2) * HID_), 16 * HID_);
      wait_tensor();
    }
#else
#pragma unroll
    for (int i = 0; i < 8; ++i) {
      int o = (t + i * 256) * 4;   // u32 index into slab
      async_copy16(W1p + (size_t)(kk / 2) * HID_ + o, (char*)Bs + o * 4);
    }
#endif
    wait_async();
    __syncthreads();

    v16bf a = load_a_frag(As + (wm * 16 + ln) * 32, g);
#pragma unroll
    for (int nt = 0; nt < 16; ++nt) {
      v16bf b = load_b_frag(Bs, 512, nh * 256 + nt * 16 + ln, g);
      acc[nt] = __builtin_amdgcn_wmma_f32_16x16x32_bf16(false, a, false, b, (short)0, acc[nt], false, false);
    }
    __syncthreads();
  }

  // GEMM1 epilogue: bias + relu -> Hs (bf16); stage packed W2 into Bs (free now)
#pragma unroll
  for (int i = 0; i < 8; ++i) {
    int o = (t + i * 256) * 4;
    async_copy16(W2p + o, (char*)Bs + o * 4);
  }
#pragma unroll
  for (int nt = 0; nt < 16; ++nt) {
    int c = nh * 256 + nt * 16 + ln;
    float bv = b1[c];
#pragma unroll
    for (int i = 0; i < 8; ++i) {
      int m = wm * 16 + i + 8 * g;
      float v = acc[nt][i] + bv;
      Hs[m * 512 + c] = (__bf16)(v > 0.0f ? v : 0.0f);
    }
  }
  wait_async();
  __syncthreads();

  // GEMM2 via WMMA: preds(64x32pad) = Hs(64x512) @ W2pad(512x32)
  {
    const int nt2 = wave >> 2;     // two 16-col tiles (cols 17..31 are zero-pad)
    v8f pacc = {};
    for (int kk = 0; kk < HID_; kk += 32) {
      v16bf a = load_a_frag(Hs + (size_t)(wm * 16 + ln) * 512 + kk, g);
      v16bf b = load_b_frag(Bs + kk * 32, 32, nt2 * 16 + ln, g);
      pacc = __builtin_amdgcn_wmma_f32_16x16x32_bf16(false, a, false, b, (short)0, pacc, false, false);
    }
    int c = nt2 * 16 + ln;
    if (c < 17) {
      float bv = b2[c];
#pragma unroll
      for (int i = 0; i < 8; ++i)
        Ps[(wm * 16 + i + 8 * g) * 20 + c] = pacc[i] + bv;
    }
  }
  __syncthreads();

  // Per-edge epilogue: mask, Jacobi eigh of symmetric 4x4, outputs
  if (t < 64) {
    int e = e0 + t;
    float pr[17];
#pragma unroll
    for (int i = 0; i < 17; ++i) pr[i] = Ps[t * 20 + i];

    int dn = sDst[t], sn = sSrc[t];
    float dx = pos[dn * 3 + 0] - pos[sn * 3 + 0];
    float dy = pos[dn * 3 + 1] - pos[sn * 3 + 1];
    float dz = pos[dn * 3 + 2] - pos[sn * 3 + 2];
    float mask = (sqrtf(dx * dx + dy * dy + dz * dz) < R_) ? 1.0f : 0.0f;

    float A[4][4];
    A[0][0] = pr[6];  A[0][1] = pr[7];  A[0][2] = pr[8];  A[0][3] = pr[9];
    A[1][0] = pr[7];  A[1][1] = pr[10]; A[1][2] = pr[11]; A[1][3] = pr[12];
    A[2][0] = pr[8];  A[2][1] = pr[11]; A[2][2] = pr[13]; A[2][3] = pr[14];
    A[3][0] = pr[9];  A[3][1] = pr[12]; A[3][2] = pr[14]; A[3][3] = pr[15];
    float V[4][4] = {{1,0,0,0},{0,1,0,0},{0,0,1,0},{0,0,0,1}};

    for (int sweep = 0; sweep < 8; ++sweep) {
      for (int pp = 0; pp < 3; ++pp) {
        for (int qq = pp + 1; qq < 4; ++qq) {
          float apq = A[pp][qq];
          if (fabsf(apq) > 1e-12f) {
            float theta = (A[qq][qq] - A[pp][pp]) / (2.0f * apq);
            float tt = (theta >= 0.0f ? 1.0f : -1.0f) /
                       (fabsf(theta) + sqrtf(theta * theta + 1.0f));
            float c = 1.0f / sqrtf(tt * tt + 1.0f);
            float s = tt * c;
            for (int k = 0; k < 4; ++k) {
              float akp = A[k][pp], akq = A[k][qq];
              A[k][pp] = c * akp - s * akq;
              A[k][qq] = s * akp + c * akq;
            }
            for (int k = 0; k < 4; ++k) {
              float apk = A[pp][k], aqk = A[qq][k];
              A[pp][k] = c * apk - s * aqk;
              A[qq][k] = s * apk + c * aqk;
            }
            for (int k = 0; k < 4; ++k) {
              float vkp = V[k][pp], vkq = V[k][qq];
              V[k][pp] = c * vkp - s * vkq;
              V[k][qq] = s * vkp + c * vkq;
            }
          }
        }
      }
    }
    int best = 0; float bestv = A[0][0];
    for (int k = 1; k < 4; ++k)
      if (A[k][k] < bestv) { bestv = A[k][k]; best = k; }

    // pos 3E | pos_var 3E | rot 4E | rot_var E | edge_index 2E | batch N | mask E
    const size_t E = (size_t)E_TOT;
    float* opos  = out;
    float* opv   = out + 3 * E;
    float* orot  = out + 6 * E;
    float* orv   = out + 10 * E;
    float* eidx  = out + 11 * E;
    float* omask = out + 13 * E + (size_t)NNODE;

    opos[e * 3 + 0] = mask * pr[0];
    opos[e * 3 + 1] = mask * pr[1];
    opos[e * 3 + 2] = mask * pr[2];
    opv[e * 3 + 0]  = mask * __expf(pr[3]);
    opv[e * 3 + 1]  = mask * __expf(pr[4]);
    opv[e * 3 + 2]  = mask * __expf(pr[5]);
#pragma unroll
    for (int r = 0; r < 4; ++r) orot[e * 4 + r] = mask * V[r][best];
    orv[e]      = mask * __expf(pr[16]);
    eidx[e]     = (float)sn;
    eidx[E + e] = (float)dn;
    omask[e]    = mask;
  }
}

// ---------------------------------------------------------------------------
__global__ void fill_batch(float* __restrict__ ob) {
  int n = blockIdx.x * 256 + threadIdx.x;
  if (n < NNODE) ob[n] = (float)(n / NN_);
}

// ---------------------------------------------------------------------------
extern "C" void kernel_launch(void* const* d_in, const int* in_sizes, int n_in,
                              void* d_out, int out_size, void* d_ws, size_t ws_size,
                              hipStream_t stream) {
  const float* img  = (const float*)d_in[0];
  const float* pos  = (const float*)d_in[1];
  // d_in[2] = rot_quat (unused by reference)
  const float* encW = (const float*)d_in[3];
  const float* encb = (const float*)d_in[4];
  const float* W1   = (const float*)d_in[5];
  const float* b1   = (const float*)d_in[6];
  const float* W2   = (const float*)d_in[7];
  const float* b2   = (const float*)d_in[8];
  float* out = (float*)d_out;

  // Workspace layout (bytes)
  char* ws = (char*)d_ws;
  __bf16*   xbf   = (__bf16*)ws;                               // 7,864,320 B
  unsigned* encWp = (unsigned*)(ws + 7864320);                 // 2,359,296 B
  unsigned* W1p   = (unsigned*)(ws + 7864320 + 2359296);       //   786,432 B
  unsigned* W2p   = (unsigned*)(ws + 7864320 + 2359296 + 786432); // 32,768 B

  pack_weights<<<512, 256, 0, stream>>>(encW, W1, W2, encWp, W1p, W2p);

  dim3 g1(D_ / 64, NNODE / 128);   // 6 x 80
  enc_gemm_relu<<<g1, 256, 0, stream>>>(img, encWp, encb, xbf);

  msg_edge_kernel<<<E_TOT / 64, 256, 0, stream>>>(xbf, pos, W1p, b1, W2p, b2, out);

  fill_batch<<<(NNODE + 255) / 256, 256, 0, stream>>>(out + (size_t)13 * E_TOT);
}
